// ConformerDiffusion_9689446219798
// MI455X (gfx1250) — matrix operand order repaired and checked by reading, not compiled, not run on hardware
//
#include <hip/hip_runtime.h>
#include <math.h>

typedef _Float16 half_t;
typedef __attribute__((ext_vector_type(16))) _Float16 v16h;
typedef __attribute__((ext_vector_type(8)))  _Float16 v8h;
typedef __attribute__((ext_vector_type(8)))  float    v8f;

#define HID   256
#define EDIMC 32
#define NL    6

__device__ __forceinline__ float siluf(float x) { return x / (1.0f + __expf(-x)); }

__device__ __forceinline__ v8f wmma16(v16h a, v16h b, v8f c) {
    return __builtin_amdgcn_wmma_f32_16x16x32_f16(false, a, false, b, (short)0, c, false, false);
}

// A fragment: 16 rows (rowBase..rowBase+15) x 32 K-slice from an LDS tile (row stride ldk halfs).
// ISA 16-bit A layout: lane m=lane&15 holds row M=m; element e: K = (e/8)*16 + (lane>>4)*8 + (e%8).
__device__ __forceinline__ v16h lds_loadA(const half_t* tile, int ldk, int rowBase, int kBase) {
    int lane = threadIdx.x & 31;
    int m = lane & 15, g = lane >> 4;
    const half_t* rp = tile + (rowBase + m) * ldk + kBase + g * 8;
    v8h lo = *(const v8h*)(rp);        // K = g*8 + e        (e=0..7)
    v8h hi = *(const v8h*)(rp + 16);   // K = 16 + g*8 + e   (e=8..15)
    v16h a;
#pragma unroll
    for (int i = 0; i < 8; ++i) { a[i] = lo[i]; a[i + 8] = hi[i]; }
    return a;
}

// B fragment: 32(K) x 16(N) from transposed f16 weights wT[N][ldk].
// ISA B layout: lane holds column N=lane&15; element e: K = (e/8)*16 + (lane>>4)*8 + (e%8).
__device__ __forceinline__ v16h glb_loadB(const half_t* wT, int ldk, int nBase, int kBase) {
    int lane = threadIdx.x & 31;
    const half_t* p = wT + (size_t)(nBase + (lane & 15)) * ldk + kBase + (lane >> 4) * 8;
    v8h lo = *(const v8h*)(p);
    v8h hi = *(const v8h*)(p + 16);
    v16h b;
#pragma unroll
    for (int i = 0; i < 8; ++i) { b[i] = lo[i]; b[i + 8] = hi[i]; }
    return b;
}

// ---------------------------------------------------------------- small utility kernels
__global__ void k_t16(const float* __restrict__ src, half_t* __restrict__ dst, int K, int Nc) {
    int i = blockIdx.x * 256 + threadIdx.x;
    if (i < K * Nc) {
        int k = i / Nc, n = i % Nc;
        dst[(size_t)n * K + k] = (half_t)src[(size_t)k * Nc + n];
    }
}
__global__ void k_cvt16(const float* __restrict__ src, half_t* __restrict__ dst, int n) {
    int i = blockIdx.x * 256 + threadIdx.x;
    if (i < n) dst[i] = (half_t)src[i];
}
__global__ void k_zerof(float* __restrict__ p, int n) {
    int i = blockIdx.x * 256 + threadIdx.x;
    if (i < n) p[i] = 0.0f;
}
__global__ void k_copyf(const float* __restrict__ s, float* __restrict__ d, int n) {
    int i = blockIdx.x * 256 + threadIdx.x;
    if (i < n) d[i] = s[i];
}

// ---------------------------------------------------------------- time embedding MLP (B=2048 rows)
__global__ void __launch_bounds__(256)
k_time(const int* __restrict__ t, const float* __restrict__ w1, const float* __restrict__ b1,
       const float* __restrict__ w2, const float* __restrict__ b2, float* __restrict__ temb) {
    __shared__ float se[128];
    __shared__ float sh1[256];
    int b = blockIdx.x, tid = threadIdx.x;
    float tv = (float)t[b];
    if (tid < 64) {
        float f = __expf((float)tid * (-logf(10000.0f) / 63.0f));
        float a = tv * f;
        se[tid] = sinf(a);
        se[tid + 64] = cosf(a);
    }
    __syncthreads();
    float a1 = b1[tid];
    for (int k = 0; k < 128; ++k) a1 += se[k] * w1[k * 256 + tid];
    sh1[tid] = siluf(a1);
    __syncthreads();
    float a2 = b2[tid];
    for (int k = 0; k < 256; ++k) a2 += sh1[k] * w2[k * 256 + tid];
    temb[(size_t)b * 256 + tid] = a2;
}

// ---------------------------------------------------------------- node init: h = atom_emb + x@Wc + bc + temb[batch]
__global__ void k_init(const float* __restrict__ x, const int* __restrict__ at,
                       const int* __restrict__ bidx, const float* __restrict__ aemb,
                       const float* __restrict__ cw, const float* __restrict__ cb,
                       const float* __restrict__ temb,
                       float* __restrict__ h, half_t* __restrict__ h16, int N) {
    size_t i = (size_t)blockIdx.x * 256 + threadIdx.x;
    if (i >= (size_t)N * HID) return;
    int n = (int)(i >> 8), c = (int)(i & 255);
    int a = at[n]; a = a < 0 ? 0 : (a > 9 ? 9 : a);
    float v = aemb[a * HID + c]
            + x[n * 3 + 0] * cw[c] + x[n * 3 + 1] * cw[HID + c] + x[n * 3 + 2] * cw[2 * HID + c]
            + cb[c] + temb[(size_t)bidx[n] * HID + c];
    h[i] = v;
    h16[i] = (half_t)v;
}

// ---------------------------------------------------------------- fused edge pipeline (one layer)
// tile: 32 edges x 256 cols.  MLP1 (K=545 decomposed) -> MLP2 -> cmlp1 -> cmlp2 -> atomic scatters.
__global__ void __launch_bounds__(256)
k_edge(const int* __restrict__ erow, const int* __restrict__ ecol,
       const float* __restrict__ x, const half_t* __restrict__ h16,
       const int* __restrict__ bond, const half_t* __restrict__ bemb16,
       const half_t* __restrict__ w1aT, const half_t* __restrict__ w1bT,
       const half_t* __restrict__ w1cT, const float* __restrict__ w1d,
       const float* __restrict__ b1,
       const half_t* __restrict__ w2T, const float* __restrict__ b2,
       const half_t* __restrict__ cw1T, const float* __restrict__ cb1,
       const float* __restrict__ cw2, const float* __restrict__ cb2,
       float* __restrict__ magg, float* __restrict__ xn, int E) {
    __shared__ __align__(16) half_t shA[32 * HID];    // h[row], later m
    __shared__ __align__(16) half_t shB[32 * HID];    // h[col], later cm1 (32x128)
    __shared__ __align__(16) half_t shM1[32 * HID];   // MLP1 output
    __shared__ __align__(16) half_t shEA[32 * EDIMC]; // edge attr
    __shared__ float shDist[32];
    __shared__ float shUnit[32 * 3];
    __shared__ int shRow[32], shCol[32], shBt[32], shVal[32];

    int tid = threadIdx.x;
    int e0 = blockIdx.x * 32;
    if (tid < 32) {
        int e = e0 + tid;
        int valid = (e < E);
        int ee = valid ? e : 0;
        int r = erow[ee], c = ecol[ee];
        shRow[tid] = r; shCol[tid] = c; shVal[tid] = valid;
        int bt = bond[ee]; bt = bt < 0 ? 0 : (bt > 4 ? 4 : bt);
        shBt[tid] = bt;
        float dx = x[r * 3 + 0] - x[c * 3 + 0];
        float dy = x[r * 3 + 1] - x[c * 3 + 1];
        float dz = x[r * 3 + 2] - x[c * 3 + 2];
        float d = fmaxf(sqrtf(dx * dx + dy * dy + dz * dz), 1e-6f);
        shDist[tid] = d;
        shUnit[tid * 3 + 0] = dx / d; shUnit[tid * 3 + 1] = dy / d; shUnit[tid * 3 + 2] = dz / d;
    }
    __syncthreads();
    for (int c4 = tid; c4 < 32 * 32; c4 += 256) {     // 32 edges x (512B rows as 32 uint4)
        int e = c4 >> 5, off = c4 & 31;
        ((uint4*)shA)[c4] = ((const uint4*)(h16 + (size_t)shRow[e] * HID))[off];
        ((uint4*)shB)[c4] = ((const uint4*)(h16 + (size_t)shCol[e] * HID))[off];
    }
    if (tid < 128) {
        int e = tid >> 2, off = tid & 3;
        ((uint4*)shEA)[tid] = ((const uint4*)(bemb16 + shBt[e] * EDIMC))[off];
    }
    __syncthreads();

    int wv = tid >> 5, lane = tid & 31;
    int g = lane >> 4, nl = lane & 15;
    int rowBase = (wv & 1) * 16;
    int nBase = (wv >> 1) * 64;
    v8f z = {0.f, 0.f, 0.f, 0.f, 0.f, 0.f, 0.f, 0.f};

    // ---- MLP1: h[row]@W1a + h[col]@W1b + ea@W1c + dist*w1d + b1, silu ----
    v8f acc[4] = {z, z, z, z};
    for (int kc = 0; kc < HID; kc += 32) {
        v16h a = lds_loadA(shA, HID, rowBase, kc);
#pragma unroll
        for (int nt = 0; nt < 4; ++nt)
            acc[nt] = wmma16(a, glb_loadB(w1aT, HID, nBase + nt * 16, kc), acc[nt]);
    }
    for (int kc = 0; kc < HID; kc += 32) {
        v16h a = lds_loadA(shB, HID, rowBase, kc);
#pragma unroll
        for (int nt = 0; nt < 4; ++nt)
            acc[nt] = wmma16(a, glb_loadB(w1bT, HID, nBase + nt * 16, kc), acc[nt]);
    }
    {
        v16h a = lds_loadA(shEA, EDIMC, rowBase, 0);
#pragma unroll
        for (int nt = 0; nt < 4; ++nt)
            acc[nt] = wmma16(a, glb_loadB(w1cT, EDIMC, nBase + nt * 16, 0), acc[nt]);
    }
#pragma unroll
    for (int nt = 0; nt < 4; ++nt) {
        int n = nBase + nt * 16 + nl;
        float bb = b1[n], wd = w1d[n];
#pragma unroll
        for (int j = 0; j < 8; ++j) {
            int rM = rowBase + j + g * 8;
            shM1[rM * HID + n] = (half_t)siluf(acc[nt][j] + shDist[rM] * wd + bb);
        }
    }
    __syncthreads();

    // ---- MLP2: silu(m1@W2 + b2); scatter-add into m_agg[col] ----
    v8f acc2[4] = {z, z, z, z};
    for (int kc = 0; kc < HID; kc += 32) {
        v16h a = lds_loadA(shM1, HID, rowBase, kc);
#pragma unroll
        for (int nt = 0; nt < 4; ++nt)
            acc2[nt] = wmma16(a, glb_loadB(w2T, HID, nBase + nt * 16, kc), acc2[nt]);
    }
#pragma unroll
    for (int nt = 0; nt < 4; ++nt) {
        int n = nBase + nt * 16 + nl;
        float bb = b2[n];
#pragma unroll
        for (int j = 0; j < 8; ++j) {
            int rM = rowBase + j + g * 8;
            float v = siluf(acc2[nt][j] + bb);
            shA[rM * HID + n] = (half_t)v;                                  // m
            if (shVal[rM]) atomicAdd(&magg[(size_t)shCol[rM] * HID + n], v);
        }
    }
    __syncthreads();

    // ---- cmlp1: silu(m@Wc1 + bc1), 32x128 ----
    v8f acc3[2] = {z, z};
    int nBaseC = (wv >> 1) * 32;
    for (int kc = 0; kc < HID; kc += 32) {
        v16h a = lds_loadA(shA, HID, rowBase, kc);
#pragma unroll
        for (int nt = 0; nt < 2; ++nt)
            acc3[nt] = wmma16(a, glb_loadB(cw1T, HID, nBaseC + nt * 16, kc), acc3[nt]);
    }
#pragma unroll
    for (int nt = 0; nt < 2; ++nt) {
        int n = nBaseC + nt * 16 + nl;
        float bb = cb1[n];
#pragma unroll
        for (int j = 0; j < 8; ++j) {
            int rM = rowBase + j + g * 8;
            shB[rM * 128 + n] = (half_t)siluf(acc3[nt][j] + bb);
        }
    }
    __syncthreads();

    // ---- cmlp2 (dot128) + coordinate scatter ----
    if (tid < 32 && shVal[tid]) {
        float s = cb2[0];
        for (int k = 0; k < 128; ++k) s += (float)shB[tid * 128 + k] * cw2[k];
        int cn = shCol[tid];
        atomicAdd(&xn[cn * 3 + 0], s * shUnit[tid * 3 + 0]);
        atomicAdd(&xn[cn * 3 + 1], s * shUnit[tid * 3 + 1]);
        atomicAdd(&xn[cn * 3 + 2], s * shUnit[tid * 3 + 2]);
    }
}

// ---------------------------------------------------------------- node MLP + residual + layernorm
__global__ void __launch_bounds__(256)
k_node(const float* __restrict__ h, const half_t* __restrict__ h16,
       const float* __restrict__ magg,
       const half_t* __restrict__ w1aT, const half_t* __restrict__ w1bT,
       const float* __restrict__ b1,
       const half_t* __restrict__ w2T, const float* __restrict__ b2,
       const float* __restrict__ lng, const float* __restrict__ lnb,
       float* __restrict__ hOut, half_t* __restrict__ h16Out, int N) {
    __shared__ __align__(16) half_t smem[3 * 32 * HID];   // 48KB
    half_t* shH = smem;
    half_t* shM = smem + 32 * HID;
    half_t* shT = smem + 2 * 32 * HID;
    float* shO = (float*)smem;                            // f32 overlay of shH+shM (32KB)

    int tid = threadIdx.x;
    int n0 = blockIdx.x * 32;
    for (int c4 = tid; c4 < 32 * 32; c4 += 256) {
        int r = c4 >> 5, off = c4 & 31;
        int nr = n0 + r; if (nr >= N) nr = N - 1;
        ((uint4*)shH)[c4] = ((const uint4*)(h16 + (size_t)nr * HID))[off];
    }
    for (int c = tid; c < 32 * HID; c += 256) {
        int r = c >> 8, k = c & 255;
        int nr = n0 + r; if (nr >= N) nr = N - 1;
        shM[c] = (half_t)magg[(size_t)nr * HID + k];
    }
    __syncthreads();

    int wv = tid >> 5, lane = tid & 31, g = lane >> 4, nl = lane & 15;
    int rowBase = (wv & 1) * 16;
    int nBase = (wv >> 1) * 64;
    v8f z = {0.f, 0.f, 0.f, 0.f, 0.f, 0.f, 0.f, 0.f};

    v8f acc[4] = {z, z, z, z};                            // K=512: [h | m_agg]
    for (int kc = 0; kc < HID; kc += 32) {
        v16h a = lds_loadA(shH, HID, rowBase, kc);
#pragma unroll
        for (int nt = 0; nt < 4; ++nt)
            acc[nt] = wmma16(a, glb_loadB(w1aT, HID, nBase + nt * 16, kc), acc[nt]);
    }
    for (int kc = 0; kc < HID; kc += 32) {
        v16h a = lds_loadA(shM, HID, rowBase, kc);
#pragma unroll
        for (int nt = 0; nt < 4; ++nt)
            acc[nt] = wmma16(a, glb_loadB(w1bT, HID, nBase + nt * 16, kc), acc[nt]);
    }
#pragma unroll
    for (int nt = 0; nt < 4; ++nt) {
        int n = nBase + nt * 16 + nl;
        float bb = b1[n];
#pragma unroll
        for (int j = 0; j < 8; ++j) {
            int rM = rowBase + j + g * 8;
            shT[rM * HID + n] = (half_t)siluf(acc[nt][j] + bb);
        }
    }
    __syncthreads();

    v8f acc2[4] = {z, z, z, z};
    for (int kc = 0; kc < HID; kc += 32) {
        v16h a = lds_loadA(shT, HID, rowBase, kc);
#pragma unroll
        for (int nt = 0; nt < 4; ++nt)
            acc2[nt] = wmma16(a, glb_loadB(w2T, HID, nBase + nt * 16, kc), acc2[nt]);
    }
#pragma unroll
    for (int nt = 0; nt < 4; ++nt) {
        int n = nBase + nt * 16 + nl;
        float bb = b2[n];
#pragma unroll
        for (int j = 0; j < 8; ++j) {
            int rM = rowBase + j + g * 8;
            int nr = n0 + rM; int nrc = nr < N ? nr : N - 1;
            shO[rM * HID + n] = acc2[nt][j] + bb + h[(size_t)nrc * HID + n];
        }
    }
    __syncthreads();

    // layernorm: 8 lanes per row, shuffle reduction inside wave32
    int r = tid >> 3, s = tid & 7;
    float sm = 0.f, sq = 0.f;
    for (int i = 0; i < 32; ++i) {
        float v = shO[r * HID + s * 32 + i];
        sm += v; sq += v * v;
    }
    for (int off = 1; off < 8; off <<= 1) {
        sm += __shfl_xor(sm, off, 32);
        sq += __shfl_xor(sq, off, 32);
    }
    float mu = sm * (1.0f / HID);
    float var = sq * (1.0f / HID) - mu * mu;
    float rs = rsqrtf(fmaxf(var, 0.f) + 1e-5f);
    int nr = n0 + r;
    if (nr < N) {
        for (int i = 0; i < 32; ++i) {
            int c = s * 32 + i;
            float v = (shO[r * HID + c] - mu) * rs * lng[c] + lnb[c];
            hOut[(size_t)nr * HID + c] = v;
            h16Out[(size_t)nr * HID + c] = (half_t)v;
        }
    }
}

// ---------------------------------------------------------------- noise prediction head
__global__ void __launch_bounds__(256)
k_head(const half_t* __restrict__ h16, const half_t* __restrict__ w1T,
       const float* __restrict__ b1, const float* __restrict__ w2,
       const float* __restrict__ b2, float* __restrict__ out, int N) {
    __shared__ __align__(16) half_t shH[32 * HID];
    __shared__ float shT[32 * HID];
    int tid = threadIdx.x;
    int n0 = blockIdx.x * 32;
    for (int c4 = tid; c4 < 32 * 32; c4 += 256) {
        int r = c4 >> 5, off = c4 & 31;
        int nr = n0 + r; if (nr >= N) nr = N - 1;
        ((uint4*)shH)[c4] = ((const uint4*)(h16 + (size_t)nr * HID))[off];
    }
    __syncthreads();

    int wv = tid >> 5, lane = tid & 31, g = lane >> 4, nl = lane & 15;
    int rowBase = (wv & 1) * 16;
    int nBase = (wv >> 1) * 64;
    v8f z = {0.f, 0.f, 0.f, 0.f, 0.f, 0.f, 0.f, 0.f};
    v8f acc[4] = {z, z, z, z};
    for (int kc = 0; kc < HID; kc += 32) {
        v16h a = lds_loadA(shH, HID, rowBase, kc);
#pragma unroll
        for (int nt = 0; nt < 4; ++nt)
            acc[nt] = wmma16(a, glb_loadB(w1T, HID, nBase + nt * 16, kc), acc[nt]);
    }
#pragma unroll
    for (int nt = 0; nt < 4; ++nt) {
        int n = nBase + nt * 16 + nl;
        float bb = b1[n];
#pragma unroll
        for (int j = 0; j < 8; ++j) {
            int rM = rowBase + j + g * 8;
            shT[rM * HID + n] = siluf(acc[nt][j] + bb);
        }
    }
    __syncthreads();
    if (tid < 96) {
        int nd = tid / 3, c = tid % 3;
        int nr = n0 + nd;
        if (nr < N) {
            float s = b2[c];
            for (int k = 0; k < HID; ++k) s += shT[nd * HID + k] * w2[k * 3 + c];
            out[(size_t)nr * 3 + c] = s;
        }
    }
}

// ================================================================ host
extern "C" void kernel_launch(void* const* d_in, const int* in_sizes, int n_in,
                              void* d_out, int out_size, void* d_ws, size_t ws_size,
                              hipStream_t stream) {
    const float* x_noisy  = (const float*)d_in[0];
    const int*   tarr     = (const int*)d_in[1];
    const int*   atom_t   = (const int*)d_in[2];
    const int*   eidx     = (const int*)d_in[3];
    const int*   bond     = (const int*)d_in[4];
    const int*   bidx     = (const int*)d_in[5];
    const float* atom_emb = (const float*)d_in[6];
    const float* bond_emb = (const float*)d_in[7];
    const float* tw1 = (const float*)d_in[8];  const float* tb1 = (const float*)d_in[9];
    const float* tw2 = (const float*)d_in[10]; const float* tb2 = (const float*)d_in[11];
    const float* cwm = (const float*)d_in[12]; const float* cbm = (const float*)d_in[13];
    const float* ew1 = (const float*)d_in[14]; const float* eb1 = (const float*)d_in[15];
    const float* ew2 = (const float*)d_in[16]; const float* eb2 = (const float*)d_in[17];
    const float* cm1 = (const float*)d_in[18]; const float* cmb1 = (const float*)d_in[19];
    const float* cm2 = (const float*)d_in[20]; const float* cmb2 = (const float*)d_in[21];
    const float* nw1 = (const float*)d_in[22]; const float* nb1 = (const float*)d_in[23];
    const float* nw2 = (const float*)d_in[24]; const float* nb2 = (const float*)d_in[25];
    const float* lng = (const float*)d_in[26]; const float* lnb = (const float*)d_in[27];
    const float* npw1 = (const float*)d_in[28]; const float* npb1 = (const float*)d_in[29];
    const float* npw2 = (const float*)d_in[30]; const float* npb2 = (const float*)d_in[31];

    int N = in_sizes[0] / 3;
    int E = in_sizes[4];
    int Bt = in_sizes[1];

    char* p = (char*)d_ws;
    auto carve = [&](size_t bytes) -> char* {
        char* q = p;
        p += (bytes + 255) & ~(size_t)255;
        return q;
    };
    float*  hA    = (float*)carve((size_t)N * HID * 4);
    float*  hB    = (float*)carve((size_t)N * HID * 4);
    half_t* h16A  = (half_t*)carve((size_t)N * HID * 2);
    half_t* h16B  = (half_t*)carve((size_t)N * HID * 2);
    float*  magg  = (float*)carve((size_t)N * HID * 4);
    float*  xA    = (float*)carve((size_t)N * 3 * 4);
    float*  xB    = (float*)carve((size_t)N * 3 * 4);
    float*  temb  = (float*)carve((size_t)Bt * HID * 4);
    half_t* w1aT  = (half_t*)carve((size_t)NL * HID * HID * 2);
    half_t* w1bT  = (half_t*)carve((size_t)NL * HID * HID * 2);
    half_t* w1cT  = (half_t*)carve((size_t)NL * HID * EDIMC * 2);
    half_t* w2T   = (half_t*)carve((size_t)NL * HID * HID * 2);
    half_t* cw1T  = (half_t*)carve((size_t)NL * 128 * HID * 2);
    half_t* nw1aT = (half_t*)carve((size_t)NL * HID * HID * 2);
    half_t* nw1bT = (half_t*)carve((size_t)NL * HID * HID * 2);
    half_t* nw2T  = (half_t*)carve((size_t)NL * HID * HID * 2);
    half_t* npw1T = (half_t*)carve((size_t)HID * HID * 2);
    half_t* bemb16 = (half_t*)carve(6 * EDIMC * 2);

    auto T16 = [&](const float* src, half_t* dst, int K, int Nc) {
        int tot = K * Nc;
        k_t16<<<dim3((tot + 255) / 256), dim3(256), 0, stream>>>(src, dst, K, Nc);
    };
    for (int l = 0; l < NL; ++l) {
        const float* e1 = ew1 + (size_t)l * 545 * HID;
        T16(e1,              w1aT + (size_t)l * HID * HID,  HID,   HID);
        T16(e1 + 256 * HID,  w1bT + (size_t)l * HID * HID,  HID,   HID);
        T16(e1 + 512 * HID,  w1cT + (size_t)l * HID * EDIMC, EDIMC, HID);
        T16(ew2 + (size_t)l * HID * HID, w2T + (size_t)l * HID * HID, HID, HID);
        T16(cm1 + (size_t)l * HID * 128, cw1T + (size_t)l * 128 * HID, HID, 128);
        const float* n1 = nw1 + (size_t)l * 512 * HID;
        T16(n1,              nw1aT + (size_t)l * HID * HID, HID, HID);
        T16(n1 + 256 * HID,  nw1bT + (size_t)l * HID * HID, HID, HID);
        T16(nw2 + (size_t)l * HID * HID, nw2T + (size_t)l * HID * HID, HID, HID);
    }
    T16(npw1, npw1T, HID, HID);
    k_cvt16<<<dim3(1), dim3(256), 0, stream>>>(bond_emb, bemb16, 6 * EDIMC);

    k_time<<<dim3(Bt), dim3(256), 0, stream>>>(tarr, tw1, tb1, tw2, tb2, temb);
    k_init<<<dim3((unsigned)(((size_t)N * HID + 255) / 256)), dim3(256), 0, stream>>>(
        x_noisy, atom_t, bidx, atom_emb, cwm, cbm, temb, hA, h16A, N);
    k_copyf<<<dim3((N * 3 + 255) / 256), dim3(256), 0, stream>>>(x_noisy, xA, N * 3);

    float* hCur = hA; float* hNext = hB;
    half_t* h16Cur = h16A; half_t* h16Next = h16B;
    float* xCur = xA; float* xNext = xB;

    for (int l = 0; l < NL; ++l) {
        k_zerof<<<dim3((unsigned)(((size_t)N * HID + 255) / 256)), dim3(256), 0, stream>>>(magg, N * HID);
        k_copyf<<<dim3((N * 3 + 255) / 256), dim3(256), 0, stream>>>(xCur, xNext, N * 3);
        k_edge<<<dim3((E + 31) / 32), dim3(256), 0, stream>>>(
            eidx, eidx + E, xCur, h16Cur, bond, bemb16,
            w1aT + (size_t)l * HID * HID, w1bT + (size_t)l * HID * HID,
            w1cT + (size_t)l * HID * EDIMC,
            ew1 + ((size_t)l * 545 + 544) * HID, eb1 + (size_t)l * HID,
            w2T + (size_t)l * HID * HID, eb2 + (size_t)l * HID,
            cw1T + (size_t)l * 128 * HID, cmb1 + (size_t)l * 128,
            cm2 + (size_t)l * 128, cmb2 + l,
            magg, xNext, E);
        k_node<<<dim3((N + 31) / 32), dim3(256), 0, stream>>>(
            hCur, h16Cur, magg,
            nw1aT + (size_t)l * HID * HID, nw1bT + (size_t)l * HID * HID, nb1 + (size_t)l * HID,
            nw2T + (size_t)l * HID * HID, nb2 + (size_t)l * HID,
            lng + (size_t)l * HID, lnb + (size_t)l * HID,
            hNext, h16Next, N);
        { float* t0 = hCur; hCur = hNext; hNext = t0; }
        { half_t* t0 = h16Cur; h16Cur = h16Next; h16Next = t0; }
        { float* t0 = xCur; xCur = xNext; xNext = t0; }
    }

    k_head<<<dim3((N + 31) / 32), dim3(256), 0, stream>>>(
        h16Cur, npw1T, npb1, npw2, npb2, (float*)d_out, N);
    (void)n_in; (void)out_size; (void)ws_size;
}
